// GPNA_2903397892151
// MI455X (gfx1250) — compile-verified, matrix operand order checked by reading, not run on hardware
//
#include <hip/hip_runtime.h>
#include <math.h>

// ---------------------------------------------------------------------------
// PNA GNN forward for MI455X (gfx1250, wave32).
// Edge GEMM algebraically decomposed into per-node GEMMs (WMMA f16->f32)
// + one edge scatter-reduce pass per layer (L2-resident atomics).
// All GEMMs share N = 50000 rows = 3125 row-tiles = 625 groups of 5 tiles.
// ---------------------------------------------------------------------------

typedef __attribute__((ext_vector_type(16))) _Float16 v16h;
typedef __attribute__((ext_vector_type(8)))  float    v8f;

#define N_NODES 50000
#define N_EDGES 800000
#define ROW_GROUPS 625      // 3125 row-tiles / 5 tiles per wave
#define R_TILES 5

// ---- ordered-uint encoding for float atomic min/max -------------------------
__device__ __forceinline__ unsigned f2ord(float f) {
  unsigned u = __float_as_uint(f);
  return (u & 0x80000000u) ? ~u : (u | 0x80000000u);
}
__device__ __forceinline__ float ord2f(unsigned u) {
  return (u & 0x80000000u) ? __uint_as_float(u & 0x7FFFFFFFu) : __uint_as_float(~u);
}

// ---- utility fill -----------------------------------------------------------
__global__ void fill_u32(unsigned* __restrict__ p, int n, unsigned v) {
  int i = blockIdx.x * blockDim.x + threadIdx.x;
  if (i < n) p[i] = v;
}

// ---- degree + avg_log -------------------------------------------------------
__global__ void deg_k(const int* __restrict__ ei, int E, float* __restrict__ deg) {
  int e = blockIdx.x * blockDim.x + threadIdx.x;
  if (e < E) atomicAdd(&deg[ei[E + e]], 1.0f);   // dst = ei[E + e]
}

__global__ void avg_log_k(const float* __restrict__ deg, int N, float* __restrict__ out) {
  int tid = blockIdx.x * blockDim.x + threadIdx.x;
  float v = (tid < N) ? logf(deg[tid] + 1.0f) : 0.0f;
  __shared__ float sm[256];
  sm[threadIdx.x] = v;
  __syncthreads();
  for (int s = 128; s > 0; s >>= 1) {
    if (threadIdx.x < s) sm[threadIdx.x] += sm[threadIdx.x + s];
    __syncthreads();
  }
  if (threadIdx.x == 0) atomicAdd(out, sm[0]);
}

// ---- edge scatter: 4 features/thread, one b128 gather, 16 atomics -----------
__global__ void edge_scatter(const int* __restrict__ ei, int E, int fsh,
                             const float* __restrict__ xj,
                             float* __restrict__ S1, float* __restrict__ S2,
                             unsigned* __restrict__ MN, unsigned* __restrict__ MX) {
  int tid = blockIdx.x * blockDim.x + threadIdx.x;
  int qsh = fsh - 2;                         // quads per row
  int total = E << qsh;
  if (tid >= total) return;
  int e = tid >> qsh;
  int j = (tid & ((1 << qsh) - 1)) << 2;     // feature base (multiple of 4)
  int s = ei[e];
  int d = ei[E + e];
  float4 v = *(const float4*)(xj + ((size_t)s << fsh) + j);
  int o = (d << fsh) + j;
#pragma unroll
  for (int t = 0; t < 4; ++t) {
    float vv = (t == 0) ? v.x : (t == 1) ? v.y : (t == 2) ? v.z : v.w;
    atomicAdd(&S1[o + t], vv);
    atomicAdd(&S2[o + t], vv * vv);
    unsigned key = f2ord(vv);
    atomicMax(&MX[o + t], key);
    atomicMin(&MN[o + t], key);
  }
}

// ---- per-node stats: overwrite S1=mean, MN=min, MX=max, S2=std; write amp ---
__global__ void node_stats(const float* __restrict__ xi, const float* __restrict__ bpre,
                           const float* __restrict__ deg, const float* __restrict__ avgsum,
                           float* __restrict__ S1, float* __restrict__ S2,
                           float* __restrict__ MN, float* __restrict__ MX,
                           float* __restrict__ amp, int N, int fsh, float fN) {
  int tid = blockIdx.x * blockDim.x + threadIdx.x;
  int total = N << fsh;
  if (tid >= total) return;
  int v = tid >> fsh;
  int k = tid & ((1 << fsh) - 1);
  float dg  = deg[v];
  float cnt = fmaxf(dg, 1.0f);
  float a   = xi[tid] + bpre[k];
  float s1  = S1[tid];
  float s2  = S2[tid];
  float m1  = (dg * a + s1) / cnt;                       // exact mean incl. deg==0
  float m2  = (dg * a * a + 2.0f * a * s1 + s2) / cnt;
  float var = m2 - m1 * m1;
  var = var > 0.0f ? var : 0.0f;
  float stdv = sqrtf(var + 1e-5f);
  bool has = dg > 0.0f;
  float mn = has ? a + ord2f(__float_as_uint(MN[tid])) : 0.0f;
  float mx = has ? a + ord2f(__float_as_uint(MX[tid])) : 0.0f;
  S1[tid] = m1;
  MN[tid] = mn;
  MX[tid] = mx;
  S2[tid] = stdv;
  if (k == 0) amp[v] = logf(cnt + 1.0f) * fN / avgsum[0];  // log(max(deg,1)+1)/avg_log
}

// ---- WMMA GEMM:  C (+)= rowcoef(row) * (A[50000xK] @ W[KxM]) (+ bias) -------
// One wave = 5 stacked 16x16 C tiles (80 rows x 16 cols); B fragment loaded
// once per k-step and reused by all 5 WMMAs. Rows never ragged (N=3125*16).
// K must be a multiple of 32. rsmode: 0=none, 1=*rs[row], 2=/rs[row].
__global__ void gemm_wmma(const float* __restrict__ A, int lda,
                          const float* __restrict__ W, int ldw,
                          float* __restrict__ C, int ldc,
                          int K, int M,
                          const float* __restrict__ rowscale, int rsmode,
                          const float* __restrict__ bias, int accum) {
  int wave = (blockIdx.x * blockDim.x + threadIdx.x) >> 5;
  int lane = threadIdx.x & 31;
  int tilesM = (M + 15) >> 4;
  if (wave >= ROW_GROUPS * tilesM) return;   // wave-uniform exit
  int trg = wave / tilesM;
  int tc  = wave % tilesM;
  int half = lane >> 4;                      // K-half per ISA 16-bit A/B layout
  int idx  = lane & 15;

  int  col  = (tc << 4) + idx;
  bool cok  = col < M;
  int  colC = cok ? col : 0;                 // clamp -> unconditional loads

  int rowbase0 = trg * (R_TILES * 16);
  const float* Alane = A + (size_t)(rowbase0 + idx) * lda;

  v8f acc[R_TILES];
#pragma unroll
  for (int r = 0; r < R_TILES; ++r) acc[r] = (v8f){};

  for (int k0 = 0; k0 < K; k0 += 32) {
    // ---- B fragment (shared by all 5 row tiles) ----
    v16h bf;
    const float* Wp = W + (size_t)(k0 + 8 * half) * ldw + colC;
#pragma unroll
    for (int i = 0; i < 8; ++i) {
      bf[i]     = (_Float16)Wp[i * ldw];
      bf[8 + i] = (_Float16)Wp[(16 + i) * ldw];
    }
#pragma unroll
    for (int r = 0; r < R_TILES; ++r) {
      const float* Ap = Alane + (size_t)r * 16 * lda + k0 + 8 * half;
      float4 q0 = *(const float4*)(Ap);        // K = k0+8h .. +3
      float4 q1 = *(const float4*)(Ap + 4);    // K = k0+8h+4 .. +7
      float4 q2 = *(const float4*)(Ap + 16);   // K = k0+16+8h .. +3
      float4 q3 = *(const float4*)(Ap + 20);   // K = k0+16+8h+4 .. +7
      v16h af;
      af[0]  = (_Float16)q0.x; af[1]  = (_Float16)q0.y;
      af[2]  = (_Float16)q0.z; af[3]  = (_Float16)q0.w;
      af[4]  = (_Float16)q1.x; af[5]  = (_Float16)q1.y;
      af[6]  = (_Float16)q1.z; af[7]  = (_Float16)q1.w;
      af[8]  = (_Float16)q2.x; af[9]  = (_Float16)q2.y;
      af[10] = (_Float16)q2.z; af[11] = (_Float16)q2.w;
      af[12] = (_Float16)q3.x; af[13] = (_Float16)q3.y;
      af[14] = (_Float16)q3.z; af[15] = (_Float16)q3.w;
      acc[r] = __builtin_amdgcn_wmma_f32_16x16x32_f16(false, af, false, bf,
                                                      (short)0, acc[r], false, false);
    }
  }

  if (cok) {
#pragma unroll
    for (int r = 0; r < R_TILES; ++r) {
      int rowb = rowbase0 + r * 16 + 8 * half;  // D: vgpr q -> M=q+8*half, N=lane&15
#pragma unroll
      for (int q = 0; q < 8; ++q) {
        int row = rowb + q;
        float v = acc[r][q];
        if (rsmode == 1)      v *= rowscale[row];
        else if (rsmode == 2) v /= rowscale[row];
        if (bias)  v += bias[col];
        size_t off = (size_t)row * ldc + col;
        if (accum) C[off] += v;
        else       C[off]  = v;
      }
    }
  }
}

// ---- ReLU in-place + per-column sum / sumsq accumulation --------------------
__global__ void relu_bn_acc(float* __restrict__ z, int total, int fomask,
                            float* __restrict__ sums, float* __restrict__ sq) {
  __shared__ float ls[128], lq[128];
  int t = threadIdx.x;
  if (t < 128) { ls[t] = 0.f; lq[t] = 0.f; }
  __syncthreads();
  for (int idx = blockIdx.x * blockDim.x + t; idx < total; idx += gridDim.x * blockDim.x) {
    float v = z[idx];
    v = v > 0.f ? v : 0.f;
    z[idx] = v;
    int c = idx & fomask;
    atomicAdd(&ls[c], v);
    atomicAdd(&lq[c], v * v);
  }
  __syncthreads();
  if (t <= fomask) { atomicAdd(&sums[t], ls[t]); atomicAdd(&sq[t], lq[t]); }
}

__global__ void bn_final(const float* __restrict__ s, const float* __restrict__ q,
                         const float* __restrict__ g, const float* __restrict__ b,
                         float* __restrict__ Ao, float* __restrict__ Bo,
                         int fo, float invN) {
  int t = threadIdx.x;
  if (t >= fo) return;
  float m   = s[t] * invN;
  float var = q[t] * invN - m * m;
  float r   = rsqrtf(var + 1e-5f);
  float a   = g[t] * r;
  Ao[t] = a;
  Bo[t] = b[t] - m * a;
}

__global__ void bn_apply(const float* __restrict__ z, const float* __restrict__ Ac,
                         const float* __restrict__ Bc, float* __restrict__ out,
                         int total, int fomask) {
  int idx = blockIdx.x * blockDim.x + threadIdx.x;
  if (idx < total) {
    int c = idx & fomask;
    out[idx] = z[idx] * Ac[c] + Bc[c];
  }
}

// ---------------------------------------------------------------------------
extern "C" void kernel_launch(void* const* d_in, const int* in_sizes, int n_in,
                              void* d_out, int out_size, void* d_ws, size_t ws_size,
                              hipStream_t stream) {
  const int N = N_NODES, E = N_EDGES;
  const float* x  = (const float*)d_in[0];
  const int*   ei = (const int*)d_in[1];
  auto P = [&](int i) { return (const float*)d_in[i]; };

  // ---- workspace layout (floats) ----
  float* w = (float*)d_ws;
  float* deg    = w; w += N;
  float* avgsum = w; w += 64;
  float* amp    = w; w += N;
  float* bnsum  = w; w += 128;
  float* bnsq   = w; w += 128;
  float* bnA    = w; w += 128;
  float* bnB    = w; w += 128;
  float* xcur   = w; w += (size_t)N * 128;
  float* xi     = w; w += (size_t)N * 128;   // reused as z (post-lin) buffer
  float* xj     = w; w += (size_t)N * 128;   // reused as outpre buffer
  float* S1     = w; w += (size_t)N * 128;
  float* S2     = w; w += (size_t)N * 128;
  float* MN     = w; w += (size_t)N * 128;
  float* MX     = w; w += (size_t)N * 128;

  auto fill = [&](void* p, int n, unsigned v) {
    fill_u32<<<(n + 255) / 256, 256, 0, stream>>>((unsigned*)p, n, v);
  };
  auto gemm = [&](const float* A, int lda, const float* W, int ldw, float* C, int ldc,
                  int k, int m, const float* rs, int rsmode,
                  const float* bias, int accum) {
    int tilesM  = (m + 15) / 16;
    int waves   = ROW_GROUPS * tilesM;
    int threads = waves * 32;
    gemm_wmma<<<(threads + 255) / 256, 256, 0, stream>>>(A, lda, W, ldw, C, ldc,
                                                         k, m, rs, rsmode, bias, accum);
  };

  // ---- degree + avg_log (shared across layers) ----
  fill(deg, N, 0u);
  fill(avgsum, 1, 0u);
  deg_k<<<(E + 255) / 256, 256, 0, stream>>>(ei, E, deg);
  avg_log_k<<<(N + 255) / 256, 256, 0, stream>>>(deg, N, avgsum);

  const int fi[4] = {128, 64, 128, 64};
  const int fo[4] = {64, 128, 64, 32};
  const float* xin = x;

  for (int l = 0; l < 4; ++l) {
    int f = fi[l], o = fo[l];
    int fsh = (f == 128) ? 7 : 6;
    const float* wpre  = P(2 + 8 * l + 0);
    const float* bpre  = P(2 + 8 * l + 1);
    const float* wpost = P(2 + 8 * l + 2);
    const float* bpost = P(2 + 8 * l + 3);
    const float* wlin  = P(2 + 8 * l + 4);
    const float* blin  = P(2 + 8 * l + 5);
    const float* gg    = P(2 + 8 * l + 6);
    const float* be    = P(2 + 8 * l + 7);

    // xi = x @ wpre[:f]  (dst/self term), xj = x @ wpre[f:] (src term)
    gemm(xin, f, wpre,                 f, xi, f, f, f, 0, 0, 0, 0);
    gemm(xin, f, wpre + (size_t)f * f, f, xj, f, f, f, 0, 0, 0, 0);

    // segment reduce xj[src] into dst rows
    fill(S1, N * f, 0u);
    fill(S2, N * f, 0u);
    fill(MN, N * f, 0xFFFFFFFFu);   // ordered-uint +identity for min
    fill(MX, N * f, 0x00000000u);   // ordered-uint -identity for max
    {
      int total = E << (fsh - 2);   // 4 features per thread
      edge_scatter<<<(total + 255) / 256, 256, 0, stream>>>(
          ei, E, fsh, xj, S1, S2, (unsigned*)MN, (unsigned*)MX);
    }
    // finalize per-node stats in-place (S1=mean, MN=min, MX=max, S2=std) + amp
    {
      int total = N << fsh;
      node_stats<<<(total + 255) / 256, 256, 0, stream>>>(
          xi, bpre, deg, avgsum, S1, S2, MN, MX, amp, N, fsh, (float)N);
    }

    // post MLP: out = [x, agg, agg*amp, agg/amp] @ wpost + bpost  (13 GEMM blocks)
    float* outp = xj;   // xj raw no longer needed
    float* zb   = xi;   // xi raw no longer needed after node_stats
    gemm(xin, f, wpost, o, outp, o, f, o, 0, 0, bpost, 0);
    const float* segs[4] = {S1, MN, MX, S2};   // [mean, min, max, std]
    for (int grp = 0; grp < 3; ++grp) {        // rsmode: 0=none, 1=*amp, 2=/amp
      for (int s = 0; s < 4; ++s) {
        size_t r0 = (size_t)f * (1 + 4 * grp + s);
        gemm(segs[s], f, wpost + r0 * o, o, outp, o, f, o, amp, grp, 0, 1);
      }
    }
    // lin
    gemm(outp, o, wlin, o, zb, o, o, o, 0, 0, blin, 0);

    // ReLU + BatchNorm (training-mode batch stats)
    fill(bnsum, o, 0u);
    fill(bnsq, o, 0u);
    relu_bn_acc<<<1024, 256, 0, stream>>>(zb, N * o, o - 1, bnsum, bnsq);
    bn_final<<<1, 128, 0, stream>>>(bnsum, bnsq, gg, be, bnA, bnB, o, 1.0f / (float)N);
    bn_apply<<<(N * o + 255) / 256, 256, 0, stream>>>(zb, bnA, bnB, xcur, N * o, o - 1);

    xin = xcur;
  }

  // classifier: out = x4 @ wc + bc   (K=32, M=10 with store-side column guard)
  gemm(xcur, 32, P(34), 10, (float*)d_out, 10, 32, 10, 0, 0, P(35), 0);
}
